// Encoder_Postnet_25383256720016
// MI455X (gfx1250) — matrix operand order repaired and checked by reading, not compiled, not run on hardware
//
#include <hip/hip_runtime.h>
#include <hip/hip_bf16.h>

// Problem constants (from the reference)
#define B_N      16
#define FRAMES_N 8192
#define TLEN_N   512
#define E_N      256
#define M_N      (B_N * FRAMES_N)   // 131072 rows
#define LDSW     257                // padded LDS row stride (floats) -> conflict-free A-frag reads

typedef __attribute__((ext_vector_type(16))) __bf16 v16bf;
typedef __attribute__((ext_vector_type(8)))  float  v8f;

// ---------------------------------------------------------------------------
// Kernel 1: parallel aligner scan (run-length automaton, one wave per batch).
// State recurrence: ind_j = (ap[j]==tp[ind_{j-1}]) ? ind_{j-1} : min(ind_{j-1}+1, 511)
// Within a constant-value segment (value v, start state ind0):
//   k = min{t>=0 : tp[clamp(ind0+t)] == v}  (k<=32 window is enough: segment len<=32)
//   ind at segment-offset t = clamp(ind0 + (k==0 ? 0 : min(t+1,k)))
// Frame 0 is forced to index 0 (reference concatenates a leading zero), so the
// automaton runs on frames 1..FRAMES-1.
// ---------------------------------------------------------------------------
__global__ __launch_bounds__(32) void align_scan_kernel(
    const int* __restrict__ ap,   // [B, FRAMES]
    const int* __restrict__ tp,   // [B, TLEN]
    int* __restrict__ inds)       // [B, FRAMES] (workspace)
{
    const int b    = blockIdx.x;
    const int lane = threadIdx.x;

    __shared__ int tps[TLEN_N];
    __shared__ int av[32];

    for (int i = lane; i < TLEN_N; i += 32) tps[i] = tp[b * TLEN_N + i];
    __syncthreads();

    const int* apb  = ap   + (size_t)b * FRAMES_N;
    int*       indb = inds + (size_t)b * FRAMES_N;

    if (lane == 0) indb[0] = 0;

    int ind = 0;   // wave-uniform automaton state
    for (int p0 = 1; p0 < FRAMES_N; p0 += 32) {
        int p = p0 + lane;
        int a = (p < FRAMES_N) ? apb[p] : -2;   // sentinel never matches tp (tp >= 1)
        av[lane] = a;
        __syncthreads();

        int aprev = (lane == 0) ? -1 : av[lane - 1];
        unsigned flags = (unsigned)__ballot(a != aprev) | 1u;  // chunk start = fresh segment (always correct)
        unsigned rem = flags;
        while (rem) {
            int s = __ffs((int)rem) - 1;
            rem &= rem - 1u;
            int e = rem ? (__ffs((int)rem) - 1) : 32;
            int v = av[s];

            // 32-lane search for first tp match at/after current state
            int idx = ind + lane; if (idx > TLEN_N - 1) idx = TLEN_N - 1;
            unsigned mhit = (unsigned)__ballot(tps[idx] == v);
            int k = mhit ? (__ffs((int)mhit) - 1) : 1000000;   // no match in 32 >= len -> pure ramp

            int len = e - s;
            int t = lane;
            if (t < len && (p0 + s + t) < FRAMES_N) {
                int adv_t = (k == 0) ? 0 : ((t + 1 < k) ? t + 1 : k);
                int indt = ind + adv_t; if (indt > TLEN_N - 1) indt = TLEN_N - 1;
                indb[p0 + s + t] = indt;
            }
            int adv = (k == 0) ? 0 : ((len < k) ? len : k);
            ind += adv; if (ind > TLEN_N - 1) ind = TLEN_N - 1;
        }
        __syncthreads();
    }
}

// ---------------------------------------------------------------------------
// Kernel 2: fused gather + PE + bf16 split-precision WMMA GEMM + epilogue.
// Block = 512 threads (16 wave32), one 32-row M-tile, each wave one 16-col N-stripe.
// out = aligner + pitch*w + b_pitch + emb_beats[beat] + (aligner+PE) @ Wᵀ + b_pos
// ---------------------------------------------------------------------------
__global__ __launch_bounds__(512) void fused_postnet_kernel(
    const float* __restrict__ enc,     // [B, TLEN, E]
    const float* __restrict__ pitch,   // [B, FRAMES]
    const int*   __restrict__ beats,   // [B, FRAMES]
    const float* __restrict__ fpw,     // [E]  fc_pitch_w
    const float* __restrict__ fpb,     // [E]  fc_pitch_b
    const float* __restrict__ Wpos,    // [E, E] fc_pos_w (row d, col e)
    const float* __restrict__ bpos,    // [E]  fc_pos_b
    const float* __restrict__ ebeats,  // [2, E]
    const int*   __restrict__ inds,    // [B, FRAMES]
    float* __restrict__ out)           // [B, FRAMES, E]
{
    __shared__ float ldsX[32 * LDSW];  // aligner + PE (GEMM input)
    __shared__ float ldsO[16 * 256];   // output staging for coalesced stores

    const int   tid = threadIdx.x;
    const int   m0  = blockIdx.x * 32;
    const float kPE = -9.210340371976184f / 256.0f;   // -ln(10000)/E

    // ---- stage X = gather(encoder_out, inds) + positional encoding ----
    for (int i = tid; i < 32 * 256; i += 512) {
        int r = i >> 8, c = i & 255;
        int m = m0 + r;
        int b = m >> 13, f = m & 8191;
        int ind = inds[b * FRAMES_N + f];
        float a = enc[((size_t)(b * TLEN_N + ind)) * E_N + c];
        float dv  = __expf((float)(c & ~1) * kPE);
        float arg = (float)f * dv;
        float pe  = (c & 1) ? __cosf(arg) : __sinf(arg);
        ldsX[r * LDSW + c] = a + pe;
    }
    __syncthreads();

    // ---- bf16 split-precision WMMA: acc += Ahi*Bhi + Alo*Bhi + Ahi*Blo ----
    const int lane = tid & 31;
    const int wave = tid >> 5;
    const int n0   = wave * 16;           // N-stripe of this wave
    const int mrow = lane & 15;           // A-frag row
    const int kb   = (lane >> 4) << 3;    // K-half offset (A) / M-half offset (C/D)

    v8f acc0 = {0.f,0.f,0.f,0.f,0.f,0.f,0.f,0.f};
    v8f acc1 = {0.f,0.f,0.f,0.f,0.f,0.f,0.f,0.f};

    for (int kc = 0; kc < E_N; kc += 32) {
        // B fragment: lane&15 = N, element j = K (same ISA 16-bit layout as A)
        v16bf bhi, blo;
        #pragma unroll
        for (int j = 0; j < 16; ++j) {
            float wv = Wpos[(n0 + mrow) * E_N + kc + kb + (j < 8 ? j : j + 8)];
            __bf16 h = (__bf16)wv;
            bhi[j] = h;
            blo[j] = (__bf16)(wv - (float)h);
        }
        #pragma unroll
        for (int mt = 0; mt < 2; ++mt) {
            v16bf ahi, alo;
            #pragma unroll
            for (int j = 0; j < 16; ++j) {
                int K = kc + kb + (j < 8 ? j : j + 8);   // ISA 16-bit A layout
                float x = ldsX[(mt * 16 + mrow) * LDSW + K];
                __bf16 h = (__bf16)x;
                ahi[j] = h;
                alo[j] = (__bf16)(x - (float)h);
            }
            if (mt == 0) {
                acc0 = __builtin_amdgcn_wmma_f32_16x16x32_bf16(false, ahi, false, bhi, (short)0, acc0, false, false);
                acc0 = __builtin_amdgcn_wmma_f32_16x16x32_bf16(false, alo, false, bhi, (short)0, acc0, false, false);
                acc0 = __builtin_amdgcn_wmma_f32_16x16x32_bf16(false, ahi, false, blo, (short)0, acc0, false, false);
            } else {
                acc1 = __builtin_amdgcn_wmma_f32_16x16x32_bf16(false, ahi, false, bhi, (short)0, acc1, false, false);
                acc1 = __builtin_amdgcn_wmma_f32_16x16x32_bf16(false, alo, false, bhi, (short)0, acc1, false, false);
                acc1 = __builtin_amdgcn_wmma_f32_16x16x32_bf16(false, ahi, false, blo, (short)0, acc1, false, false);
            }
        }
    }

    // ---- epilogue: C/D layout VGPR r -> M = r + kb, N = lane&15 ----
    const int   d    = n0 + (lane & 15);
    const float posb = bpos[d];
    const float pw   = fpw[d];
    const float pb   = fpb[d];
    const float e0   = ebeats[d];
    const float e1   = ebeats[E_N + d];
    const float divd = __expf((float)(d & ~1) * kPE);

    #pragma unroll
    for (int mt = 0; mt < 2; ++mt) {
        #pragma unroll
        for (int r = 0; r < 8; ++r) {
            int r16 = r + kb;                 // row within the 16-row tile
            int row_local = mt * 16 + r16;
            int m = m0 + row_local;
            int b = m >> 13, f = m & 8191;
            float arg = (float)f * divd;
            float pe  = (d & 1) ? __cosf(arg) : __sinf(arg);
            float al  = ldsX[row_local * LDSW + d] - pe;   // recover aligner_out
            float p   = pitch[b * FRAMES_N + f];
            int   bt  = beats[b * FRAMES_N + f];
            float accv = (mt == 0) ? acc0[r] : acc1[r];
            float v = accv + posb + al + p * pw + pb + (bt ? e1 : e0);
            ldsO[r16 * 256 + d] = v;
        }
        __syncthreads();
        // coalesced b128 stores: 16 rows x 256 floats
        for (int i = tid; i < 16 * 64; i += 512) {
            int row = i >> 6, c4 = i & 63;
            int m = m0 + mt * 16 + row;
            float4 vv = ((const float4*)(ldsO + row * 256))[c4];
            ((float4*)(out + (size_t)m * E_N))[c4] = vv;
        }
        __syncthreads();
    }
}

// ---------------------------------------------------------------------------
extern "C" void kernel_launch(void* const* d_in, const int* in_sizes, int n_in,
                              void* d_out, int out_size, void* d_ws, size_t ws_size,
                              hipStream_t stream) {
    (void)in_sizes; (void)n_in; (void)out_size; (void)ws_size;

    const float* enc   = (const float*)d_in[0];   // encoder_out
    const int*   ap    = (const int*)  d_in[1];   // align_phone
    const int*   tp    = (const int*)  d_in[2];   // text_phone
    const float* pitch = (const float*)d_in[3];
    const int*   beats = (const int*)  d_in[4];
    const float* fpw   = (const float*)d_in[5];   // fc_pitch_w [E,1]
    const float* fpb   = (const float*)d_in[6];   // fc_pitch_b [E]
    const float* Wpos  = (const float*)d_in[7];   // fc_pos_w [E,E]
    const float* bpos  = (const float*)d_in[8];   // fc_pos_b [E]
    const float* ebeat = (const float*)d_in[9];   // emb_beats [2,E]
    float* out = (float*)d_out;

    int* inds = (int*)d_ws;   // [B, FRAMES] int32 (512 KB)

    align_scan_kernel<<<B_N, 32, 0, stream>>>(ap, tp, inds);
    fused_postnet_kernel<<<M_N / 32, 512, 0, stream>>>(
        enc, pitch, beats, fpw, fpb, Wpos, bpos, ebeat, inds, out);
}
